// HausdorffBinaryLoss_26027501814216
// MI455X (gfx1250) — compile-verified, matrix operand order checked by reading, not run on hardware
//
#include <hip/hip_runtime.h>
#include <hip/hip_bf16.h>

typedef __attribute__((ext_vector_type(16))) _Float16 v16h;
typedef __attribute__((ext_vector_type(8)))  float    v8f;

#define HB 4
#define HH 256
#define HW 256
#define IMG (HH * HW)          // 65536
#define INITD 20000            // > max in-column distance, fits u16 after +256

// ---------------------------------------------------------------------------
// Kernel 1: vertical 1-D EDT (forward/backward scans along H), per column.
// grid = 16 blocks: blockIdx.x = b*4 + m  (m: 0=pc,1=~pc,2=gt,3=~gt)
// g (u16) layout: g[(b*4+m)*IMG + i*HW + j]
// ---------------------------------------------------------------------------
__global__ void hd_edt_vertical(const float* __restrict__ net,
                                const float* __restrict__ tgt,
                                unsigned short* __restrict__ g) {
    const int mb = blockIdx.x;
    const int m  = mb & 3;
    const int b  = mb >> 2;
    const int j  = threadIdx.x;

    const float* src = (m < 2) ? net : tgt;
    const float  thr = (m < 2) ? 0.0f : 0.5f;   // sigmoid(x)>0.5 <=> x>0
    const unsigned inv = (unsigned)(m & 1);

    unsigned short* gimg = g + (size_t)mb * IMG;
    const float* simg = src + (size_t)b * IMG;

    unsigned bits[8] = {0u,0u,0u,0u,0u,0u,0u,0u};

    // forward scan: distance to nearest mask==false pixel above
    int d = INITD;
    for (int i = 0; i < HH; ++i) {
        float x = simg[i * HW + j];
        unsigned on = ((x > thr) ? 1u : 0u) ^ inv;   // mask value
        bits[i >> 5] |= on << (i & 31);
        d = on ? (d + 1) : 0;
        gimg[i * HW + j] = (unsigned short)d;
    }
    // backward scan + min-combine (same-thread store->load, same address: in order)
    d = INITD;
    for (int i = HH - 1; i >= 0; --i) {
        unsigned on = (bits[i >> 5] >> (i & 31)) & 1u;
        d = on ? (d + 1) : 0;
        int f = (int)gimg[i * HW + j];
        int gm = (d < f) ? d : f;
        gimg[i * HW + j] = (unsigned short)gm;
    }
}

// ---------------------------------------------------------------------------
// Kernel 2: horizontal lower-envelope min-plus + per-pixel loss + WMMA reduce.
// grid = B*H = 1024 blocks, 256 threads (one output column each).
// ---------------------------------------------------------------------------
__global__ void hd_envelope_loss(const float* __restrict__ net,
                                 const float* __restrict__ tgt,
                                 const unsigned short* __restrict__ g,
                                 float* __restrict__ partials) {
    const int bi = blockIdx.x;
    const int b  = bi >> 8;
    const int i  = bi & 255;
    const int j  = threadIdx.x;

    __shared__ float g2s[4][HW];
    __shared__ float red[HW];

    // load this row's vertical distances (all 4 masks), squared, into LDS
    #pragma unroll
    for (int m = 0; m < 4; ++m) {
        float gv = (float)g[(size_t)(b * 4 + m) * IMG + i * HW + j];
        g2s[m][j] = gv * gv;
    }
    __syncthreads();

    // min-plus envelope: D2[j] = min_k g2[k] + (j-k)^2, (j-k)^2 shared by all masks
    const float jf = (float)j;
    float d2a = 1e30f, d2b = 1e30f, d2c = 1e30f, d2d = 1e30f;
    #pragma unroll 8
    for (int k = 0; k < HW; ++k) {
        float t  = jf - (float)k;
        float t2 = t * t;
        d2a = fminf(d2a, g2s[0][k] + t2);
        d2b = fminf(d2b, g2s[1][k] + t2);
        d2c = fminf(d2c, g2s[2][k] + t2);
        d2d = fminf(d2d, g2s[3][k] + t2);
    }

    const float pcd = sqrtf(d2a) + sqrtf(d2b);   // edt(mask) + edt(~mask)
    const float gtd = sqrtf(d2c) + sqrtf(d2d);

    const size_t idx = (size_t)b * IMG + i * HW + j;
    const float x  = net[idx];
    const float pc = 1.0f / (1.0f + expf(-x));
    const float gt = tgt[idx];
    const float e  = gt - pc;
    // pre-scale by 1/OMEGA so f16 conversion in the WMMA reduce cannot overflow
    const float val = (e * e) * (pcd * pcd + gtd * gtd) * (1.0f / 65536.0f);

    red[j] = val;
    __syncthreads();

    // --- WMMA ones-reduction: D = ones(16x32) x B(32x16) + 0 sums all B slots
    //     into each row of D; sum of row 0's 16 columns = total. Executed
    //     uniformly by every wave (EXEC must be all-ones for WMMA).
    const int lane = j & 31;
    v16h av, bv;
    #pragma unroll
    for (int t = 0; t < 16; ++t) {
        av[t] = (_Float16)1.0f;
        float v = (t < 8) ? red[t * 32 + lane] : 0.0f;  // bijection: slot = t*32+lane
        bv[t] = (_Float16)v;
    }
    v8f cv = {};
    cv = __builtin_amdgcn_wmma_f32_16x16x32_f16(
            /*neg_a=*/false, av, /*neg_b=*/false, bv,
            /*c_mod=*/(short)0, cv, /*reuse_a=*/false, /*reuse_b=*/false);
    __syncthreads();

    // c VGPR0: lanes 0..15 hold D[0, n=0..15]
    if (j < 16) red[j] = cv[0];
    __syncthreads();
    if (j == 0) {
        float s = 0.0f;
        #pragma unroll
        for (int t = 0; t < 16; ++t) s += red[t];
        partials[bi] = s;
    }
}

// ---------------------------------------------------------------------------
// Kernel 3: deterministic final reduction of 1024 partials -> scalar mean.
// ---------------------------------------------------------------------------
__global__ void hd_finalize(const float* __restrict__ partials,
                            float* __restrict__ out) {
    __shared__ float s[256];
    float t = 0.0f;
    for (int k = threadIdx.x; k < HB * HH; k += 256) t += partials[k];
    s[threadIdx.x] = t;
    __syncthreads();
    for (int off = 128; off > 0; off >>= 1) {
        if (threadIdx.x < off) s[threadIdx.x] += s[threadIdx.x + off];
        __syncthreads();
    }
    if (threadIdx.x == 0)
        out[0] = s[0] * (1.0f / (float)(HB * HH * HW));   // mean; 1/OMEGA applied per-pixel
}

// ---------------------------------------------------------------------------
extern "C" void kernel_launch(void* const* d_in, const int* in_sizes, int n_in,
                              void* d_out, int out_size, void* d_ws, size_t ws_size,
                              hipStream_t stream) {
    const float* net = (const float*)d_in[0];   // [4,1,256,256] f32
    const float* tgt = (const float*)d_in[1];   // [4,1,256,256] f32
    float* out = (float*)d_out;                 // scalar f32

    // workspace: 2MB of u16 vertical distances + 4KB partials
    unsigned short* g   = (unsigned short*)d_ws;
    float* partials     = (float*)((char*)d_ws + (size_t)(2u << 20));

    hd_edt_vertical  <<<16,          256, 0, stream>>>(net, tgt, g);
    hd_envelope_loss <<<HB * HH,     256, 0, stream>>>(net, tgt, g, partials);
    hd_finalize      <<<1,           256, 0, stream>>>(partials, out);
}
